// MultiHeadAttention_4861902979147
// MI455X (gfx1250) — matrix operand order repaired and checked by reading, not compiled, not run on hardware
//
#include <hip/hip_runtime.h>

// ---------------------------------------------------------------------------
// MI455X (gfx1250) fused MHA forward, bf16 WMMA pipeline, fp32 accumulate.
// 32M x 64N wave tiles: B fragments reused across two A fragments to cut
// VMEM issue per WMMA (loop is issue-bound, all data L2-resident).
// ---------------------------------------------------------------------------

typedef __attribute__((ext_vector_type(16))) __bf16 v16bf;
typedef __attribute__((ext_vector_type(8)))  __bf16 v8bf;
typedef __attribute__((ext_vector_type(8)))  float  v8f;

union BF16x16 { v16bf v; v8bf h[2]; };

#define B_   4
#define S_   2048
#define D_   768
#define H_   12
#define HD_  64
#define TOK  (B_ * S_)          // 8192

// ---------------------------------------------------------------------------
// fp32 -> bf16 conversion (grid-stride)
// ---------------------------------------------------------------------------
__global__ void cvt_bf16(const float* __restrict__ src, __bf16* __restrict__ dst, int n) {
    int i = blockIdx.x * blockDim.x + threadIdx.x;
    int stride = gridDim.x * blockDim.x;
    for (; i < n; i += stride) dst[i] = (__bf16)src[i];
}

// ---------------------------------------------------------------------------
// QKV projection: out[m,n] = sum_k xb[m,k] * wqkv[n,k] + b_qkv[n]
// Wave tile: 32(M) x 64(N), K stepped by 32. 8 WMMA per k-step from
// 2 A-frags + 4 B-frags (12 b128 loads).
// Epilogue scatters Q,K as [B,H,S,Hd] bf16 and V transposed [B,H,Hd,S] bf16.
// Softmax scale 1/sqrt(Hd)=0.125 folded into Q.
// ---------------------------------------------------------------------------
__global__ void __launch_bounds__(256)
qkv_gemm(const __bf16* __restrict__ xb, const __bf16* __restrict__ wb,
         const float* __restrict__ bias,
         __bf16* __restrict__ q, __bf16* __restrict__ kk, __bf16* __restrict__ vt) {
    const int lane = threadIdx.x & 31;
    const int wave = threadIdx.x >> 5;
    const int tile = blockIdx.x * 8 + wave;          // 256*36 = 9216 tiles
    const int NT   = (3 * D_) / 64;                  // 36
    const int mBase = (tile / NT) * 32;
    const int nBase = (tile % NT) * 64;

    const int half  = lane >> 4;
    const int ln16  = lane & 15;
    const int abase = half * 8;                      // A-frag K-chunk base (ISA layout)

    v8f acc[2][4] = {};
    const __bf16* arow0 = xb + (size_t)(mBase + ln16) * D_;
    const __bf16* arow1 = arow0 + (size_t)16 * D_;

    for (int k0 = 0; k0 < D_; k0 += 32) {
        __builtin_prefetch(arow0 + k0 + 128, 0, 1);  // global_prefetch_b8
        BF16x16 a[2];
        a[0].h[0] = *(const v8bf*)(arow0 + k0 + abase);
        a[0].h[1] = *(const v8bf*)(arow0 + k0 + abase + 16);
        a[1].h[0] = *(const v8bf*)(arow1 + k0 + abase);
        a[1].h[1] = *(const v8bf*)(arow1 + k0 + abase + 16);
#pragma unroll
        for (int i = 0; i < 4; ++i) {
            const __bf16* brow = wb + (size_t)(nBase + i * 16 + ln16) * D_ + k0 + half * 8;
            BF16x16 b;
            b.h[0] = *(const v8bf*)(brow);
            b.h[1] = *(const v8bf*)(brow + 16);
#pragma unroll
            for (int u = 0; u < 2; ++u)
                acc[u][i] = __builtin_amdgcn_wmma_f32_16x16x32_bf16(
                    false, a[u].v, false, b.v, (short)0, acc[u][i], false, false);
        }
    }

#pragma unroll
    for (int u = 0; u < 2; ++u) {
#pragma unroll
        for (int i = 0; i < 4; ++i) {
            const int n = nBase + i * 16 + ln16;
            const int c = n / D_;                 // 0=Q 1=K 2=V
            const int r = n % D_;
            const int h = r / HD_;
            const int d = r % HD_;
            const float bv = bias[n];
#pragma unroll
            for (int j = 0; j < 8; ++j) {
                const int m = mBase + u * 16 + j + half * 8;   // C-matrix M striping
                const int b = m >> 11;                         // /2048
                const int s = m & (S_ - 1);
                const float val = acc[u][i][j] + bv;
                const size_t hb = (size_t)(b * H_ + h);
                if (c == 0)       q [(hb * S_ + s) * HD_ + d] = (__bf16)(val * 0.125f);
                else if (c == 1)  kk[(hb * S_ + s) * HD_ + d] = (__bf16)val;
                else              vt[(hb * HD_ + d) * S_ + s] = (__bf16)val;   // transposed
            }
        }
    }
}

// ---------------------------------------------------------------------------
// Flash attention: one wave per 32-row Q tile of one (b,h). K/V fragments
// reused across both 16-row Q subtiles. Online softmax; P re-layout
// (C-frag -> A-frag) through LDS.
// ---------------------------------------------------------------------------
__global__ void __launch_bounds__(256)
attn_kernel(const __bf16* __restrict__ q, const __bf16* __restrict__ kk,
            const __bf16* __restrict__ vt, __bf16* __restrict__ ao) {
    __shared__ __bf16 pbuf[8][2][16 * 32];           // 16 KB

    const int lane = threadIdx.x & 31;
    const int wave = threadIdx.x >> 5;
    const int half = lane >> 4;
    const int ln16 = lane & 15;
    const int abase = half * 8;

    const int bh    = blockIdx.x >> 3;               // 48 (b,h) pairs, 8 blocks each
    const int qTile = (blockIdx.x & 7) * 8 + wave;   // 0..63
    const int qBase = qTile * 32;

    const __bf16* qp = q  + ((size_t)bh * S_ + qBase) * HD_;
    const __bf16* kp = kk + (size_t)bh * S_ * HD_;
    const __bf16* vp = vt + (size_t)bh * HD_ * S_;

    // Q A-fragments: 2 subtiles x k-chunks {0..31, 32..63}
    BF16x16 aq[2][2];
#pragma unroll
    for (int u = 0; u < 2; ++u) {
        const __bf16* qrow = qp + (size_t)(u * 16 + ln16) * HD_;
        aq[u][0].h[0] = *(const v8bf*)(qrow + abase);
        aq[u][0].h[1] = *(const v8bf*)(qrow + abase + 16);
        aq[u][1].h[0] = *(const v8bf*)(qrow + 32 + abase);
        aq[u][1].h[1] = *(const v8bf*)(qrow + 32 + abase + 16);
    }

    float mrun[2][8], lrun[2][8];
    v8f o[2][4] = {};
#pragma unroll
    for (int u = 0; u < 2; ++u)
#pragma unroll
        for (int j = 0; j < 8; ++j) { mrun[u][j] = -1e30f; lrun[u][j] = 0.f; }

    for (int kb = 0; kb < S_; kb += 32) {
        // ---- S = Q @ K^T for 32 keys: K-block fragments shared by both subtiles
        v8f s[2][2] = {};
#pragma unroll
        for (int t = 0; t < 2; ++t) {
            const __bf16* krow = kp + (size_t)(kb + t * 16 + ln16) * HD_ + half * 8;
            BF16x16 b0, b1;
            b0.h[0] = *(const v8bf*)(krow);
            b0.h[1] = *(const v8bf*)(krow + 16);
            b1.h[0] = *(const v8bf*)(krow + 32);
            b1.h[1] = *(const v8bf*)(krow + 48);
#pragma unroll
            for (int u = 0; u < 2; ++u) {
                s[u][t] = __builtin_amdgcn_wmma_f32_16x16x32_bf16(
                    false, aq[u][0].v, false, b0.v, (short)0, s[u][t], false, false);
                s[u][t] = __builtin_amdgcn_wmma_f32_16x16x32_bf16(
                    false, aq[u][1].v, false, b1.v, (short)0, s[u][t], false, false);
            }
        }

        // ---- online softmax: row stats via 16-lane shfl_xor reductions ----
#pragma unroll
        for (int u = 0; u < 2; ++u) {
            float p0[8], p1[8];
#pragma unroll
            for (int j = 0; j < 8; ++j) {
                float v0 = s[u][0][j], v1 = s[u][1][j];
                float mx = fmaxf(v0, v1);
#pragma unroll
                for (int off = 1; off < 16; off <<= 1)
                    mx = fmaxf(mx, __shfl_xor(mx, off, 32));
                const float mnew = fmaxf(mrun[u][j], mx);
                const float corr = __expf(mrun[u][j] - mnew);
                const float e0 = __expf(v0 - mnew);
                const float e1 = __expf(v1 - mnew);
                float rs = e0 + e1;
#pragma unroll
                for (int off = 1; off < 16; off <<= 1)
                    rs += __shfl_xor(rs, off, 32);
                lrun[u][j] = lrun[u][j] * corr + rs;
                mrun[u][j] = mnew;
                p0[j] = e0; p1[j] = e1;
#pragma unroll
                for (int i = 0; i < 4; ++i) o[u][i][j] *= corr;
            }
            // ---- P: C-fragment layout -> row-major LDS tile ----
            __bf16* pb = pbuf[wave][u];
#pragma unroll
            for (int j = 0; j < 8; ++j) {
                const int row = j + half * 8;
                pb[row * 32 + ln16]      = (__bf16)p0[j];
                pb[row * 32 + 16 + ln16] = (__bf16)p1[j];
            }
        }
        __syncthreads();
        BF16x16 pf[2];
#pragma unroll
        for (int u = 0; u < 2; ++u) {
            const __bf16* pr = pbuf[wave][u] + ln16 * 32;
            pf[u].h[0] = *(const v8bf*)(pr + abase);
            pf[u].h[1] = *(const v8bf*)(pr + abase + 16);
        }
        __syncthreads();

        // ---- O += P @ V : V^T fragments shared by both subtiles ----
#pragma unroll
        for (int i = 0; i < 4; ++i) {
            const __bf16* vrow = vp + (size_t)(i * 16 + ln16) * S_ + kb + half * 8;
            BF16x16 bv;
            bv.h[0] = *(const v8bf*)(vrow);
            bv.h[1] = *(const v8bf*)(vrow + 16);
#pragma unroll
            for (int u = 0; u < 2; ++u)
                o[u][i] = __builtin_amdgcn_wmma_f32_16x16x32_bf16(
                    false, pf[u].v, false, bv.v, (short)0, o[u][i], false, false);
        }
    }

    // ---- epilogue: normalize and store as [B,S,D] bf16 for the proj GEMM ----
    const int h = bh % H_;
    const int b = bh / H_;
#pragma unroll
    for (int u = 0; u < 2; ++u) {
#pragma unroll
        for (int i = 0; i < 4; ++i) {
            const int d = i * 16 + ln16;
#pragma unroll
            for (int j = 0; j < 8; ++j) {
                const int srow = qBase + u * 16 + j + half * 8;
                const float val = o[u][i][j] / lrun[u][j];
                ao[((size_t)(b * S_ + srow)) * D_ + h * HD_ + d] = (__bf16)val;
            }
        }
    }
}

// ---------------------------------------------------------------------------
// Output projection: out[m,n] = sum_k ao[m,k] * wproj[n,k] + b_proj[n], fp32 out
// Wave tile: 32(M) x 64(N).
// ---------------------------------------------------------------------------
__global__ void __launch_bounds__(256)
proj_gemm(const __bf16* __restrict__ ao, const __bf16* __restrict__ wb,
          const float* __restrict__ bias, float* __restrict__ out) {
    const int lane = threadIdx.x & 31;
    const int wave = threadIdx.x >> 5;
    const int tile = blockIdx.x * 8 + wave;          // 256*12 = 3072 tiles
    const int NT   = D_ / 64;                        // 12
    const int mBase = (tile / NT) * 32;
    const int nBase = (tile % NT) * 64;

    const int half  = lane >> 4;
    const int ln16  = lane & 15;
    const int abase = half * 8;

    v8f acc[2][4] = {};
    const __bf16* arow0 = ao + (size_t)(mBase + ln16) * D_;
    const __bf16* arow1 = arow0 + (size_t)16 * D_;

    for (int k0 = 0; k0 < D_; k0 += 32) {
        __builtin_prefetch(arow0 + k0 + 128, 0, 1);
        BF16x16 a[2];
        a[0].h[0] = *(const v8bf*)(arow0 + k0 + abase);
        a[0].h[1] = *(const v8bf*)(arow0 + k0 + abase + 16);
        a[1].h[0] = *(const v8bf*)(arow1 + k0 + abase);
        a[1].h[1] = *(const v8bf*)(arow1 + k0 + abase + 16);
#pragma unroll
        for (int i = 0; i < 4; ++i) {
            const __bf16* brow = wb + (size_t)(nBase + i * 16 + ln16) * D_ + k0 + half * 8;
            BF16x16 b;
            b.h[0] = *(const v8bf*)(brow);
            b.h[1] = *(const v8bf*)(brow + 16);
#pragma unroll
            for (int u = 0; u < 2; ++u)
                acc[u][i] = __builtin_amdgcn_wmma_f32_16x16x32_bf16(
                    false, a[u].v, false, b.v, (short)0, acc[u][i], false, false);
        }
    }

#pragma unroll
    for (int u = 0; u < 2; ++u) {
#pragma unroll
        for (int i = 0; i < 4; ++i) {
            const int n = nBase + i * 16 + ln16;
            const float bv = bias[n];
#pragma unroll
            for (int j = 0; j < 8; ++j) {
                const int m = mBase + u * 16 + j + half * 8;
                out[(size_t)m * D_ + n] = acc[u][i][j] + bv;
            }
        }
    }
}

// ---------------------------------------------------------------------------
// Launch
// ---------------------------------------------------------------------------
extern "C" void kernel_launch(void* const* d_in, const int* in_sizes, int n_in,
                              void* d_out, int out_size, void* d_ws, size_t ws_size,
                              hipStream_t stream) {
    (void)in_sizes; (void)n_in; (void)out_size; (void)ws_size;

    const float* x      = (const float*)d_in[0];
    const float* w_qkv  = (const float*)d_in[1];
    const float* b_qkv  = (const float*)d_in[2];
    const float* w_proj = (const float*)d_in[3];
    const float* b_proj = (const float*)d_in[4];
    float* out = (float*)d_out;

    // workspace carve-out (bf16 buffers), all sizes 256B-aligned
    char* ws = (char*)d_ws;
    const size_t SZ_XB   = (size_t)TOK * D_ * 2;          // 12,582,912
    const size_t SZ_WQKV = (size_t)3 * D_ * D_ * 2;       //  3,538,944
    const size_t SZ_WP   = (size_t)D_ * D_ * 2;           //  1,179,648
    const size_t SZ_QKV  = (size_t)TOK * D_ * 2;          // 12,582,912 each

    __bf16* xb    = (__bf16*)(ws);
    __bf16* wqkvb = (__bf16*)(ws + SZ_XB);
    __bf16* wpb   = (__bf16*)(ws + SZ_XB + SZ_WQKV);
    __bf16* qb    = (__bf16*)(ws + SZ_XB + SZ_WQKV + SZ_WP);
    __bf16* kb    = (__bf16*)(ws + SZ_XB + SZ_WQKV + SZ_WP + SZ_QKV);
    __bf16* vtb   = (__bf16*)(ws + SZ_XB + SZ_WQKV + SZ_WP + 2 * SZ_QKV);
    __bf16* aob   = (__bf16*)(ws + SZ_XB + SZ_WQKV + SZ_WP + 3 * SZ_QKV);

    cvt_bf16<<<512, 256, 0, stream>>>(x,      xb,    TOK * D_);
    cvt_bf16<<<512, 256, 0, stream>>>(w_qkv,  wqkvb, 3 * D_ * D_);
    cvt_bf16<<<512, 256, 0, stream>>>(w_proj, wpb,   D_ * D_);

    qkv_gemm   <<<1152, 256, 0, stream>>>(xb, wqkvb, b_qkv, qb, kb, vtb);
    attn_kernel<<<384,  256, 0, stream>>>(qb, kb, vtb, aob);
    proj_gemm  <<<384,  256, 0, stream>>>(aob, wpb, b_proj, out);
}
